// BiAttnModel_19980187861850
// MI455X (gfx1250) — matrix operand order repaired and checked by reading, not compiled, not run on hardware
//
#include <hip/hip_runtime.h>
#include <cstddef>
#include <cstdint>

#define BATCH 8
#define SEQ   2048
#define DIM   256

#define PITCH  264   // ushort pitch for [*][256] bf16 tiles (+pad, 16B-aligned rows)
#define PPITCH 72    // ushort pitch for [*][64]  bf16 tiles (+pad, 16B-aligned rows)

#define STATS_SMEM ((128 * PITCH + 64 * PITCH) * 2)
#define ATTN_SMEM  ((128 * PITCH + 64 * PITCH + DIM * PPITCH + 128 * PPITCH) * 2)

typedef __attribute__((ext_vector_type(16))) __bf16        v16bf;
typedef __attribute__((ext_vector_type(8)))  float         v8f;
typedef __attribute__((ext_vector_type(4)))  unsigned int  v4u;

union FragU { v16bf b; v4u q[2]; };

// native f32 -> bf16 convert (RTE), stored as raw bits for LDS staging
__device__ __forceinline__ unsigned short f2bf(float f) {
  return __builtin_bit_cast(unsigned short, (__bf16)f);
}

// order-preserving float<->uint encoding so atomicMax(uint) implements float max
__device__ __forceinline__ unsigned int fenc(float f) {
  unsigned int u = __float_as_uint(f);
  return (u & 0x80000000u) ? ~u : (u | 0x80000000u);
}
__device__ __forceinline__ float fdec(unsigned int u) {
  unsigned int v = (u & 0x80000000u) ? (u & 0x7FFFFFFFu) : ~u;
  return __uint_as_float(v);
}

// A fragment (16x32 bf16, row-major [m][k] in LDS): lane m=l&15, h=l>>4 holds
// K runs [h*8, h*8+7] and [16+h*8, 16+h*8+7].
__device__ __forceinline__ v16bf load_frag_a(const unsigned short* base, int pitch,
                                             int mBase, int kBase) {
  const int lane = threadIdx.x & 31;
  const int m = lane & 15, h = lane >> 4;
  const unsigned short* p = base + (mBase + m) * pitch + kBase + h * 8;
  FragU f;
  f.q[0] = *(const v4u*)(p);
  f.q[1] = *(const v4u*)(p + 16);
  return f.b;
}

// B fragment (32x16 bf16): lane n=l&15 holds K runs [h*8, h*8+7] and [16+h*8, 16+h*8+7]; LDS tile [n][k].
__device__ __forceinline__ v16bf load_frag_b(const unsigned short* base, int pitch,
                                             int nBase, int kBase) {
  const int lane = threadIdx.x & 31;
  const int n = lane & 15, h = lane >> 4;
  const unsigned short* p = base + (nBase + n) * pitch + kBase + h * 8;
  FragU f;
  f.q[0] = *(const v4u*)(p);
  f.q[1] = *(const v4u*)(p + 16);
  return f.b;
}

__device__ __forceinline__ v8f wmma_bf16(v16bf a, v16bf b, v8f c) {
  return __builtin_amdgcn_wmma_f32_16x16x32_bf16(false, a, false, b, (short)0, c,
                                                 false, false);
}

__global__ void init_ws(unsigned int* ws, size_t n) {
  size_t i = (size_t)blockIdx.x * blockDim.x + threadIdx.x;
  if (i < n) ws[i] = 0u;   // 0u is identity for fenc-max and for float-add
}

// Computes a 128x64 tile of M = f1 @ f2^T with WMMA (8 waves x 16 rows), then:
//   mode 0: atomic float-max into colmaxE[t] (over rows) and rowmaxE[s] (over cols)
//   mode 1: atomicAdd of exp(M - max) into colsum[t] and rowsum[s]
__global__ void __launch_bounds__(256) stats_kernel(
    const float* __restrict__ f1, const float* __restrict__ f2,
    unsigned int* __restrict__ colmaxE, unsigned int* __restrict__ rowmaxE,
    float* __restrict__ colsum, float* __restrict__ rowsum, int mode)
{
  extern __shared__ unsigned short sm[];
  unsigned short* Qs = sm;                  // [128][PITCH] bf16 of f1 strip
  unsigned short* Ks = sm + 128 * PITCH;    // [64][PITCH]  bf16 of f2 strip

  const int b  = blockIdx.z;
  const int s0 = blockIdx.x * 128;
  const int t0 = blockIdx.y * 64;
  const size_t bS = (size_t)b * SEQ;
  const float* F1 = f1 + bS * DIM;
  const float* F2 = f2 + bS * DIM;

  for (int i = threadIdx.x; i < 128 * 64; i += 256) {
    int r = i >> 6, c = (i & 63) * 4;
    float4 q = *(const float4*)(F1 + (size_t)(s0 + r) * DIM + c);
    unsigned short* dq = Qs + r * PITCH + c;
    dq[0] = f2bf(q.x); dq[1] = f2bf(q.y); dq[2] = f2bf(q.z); dq[3] = f2bf(q.w);
  }
  for (int i = threadIdx.x; i < 64 * 64; i += 256) {
    int r = i >> 6, c = (i & 63) * 4;
    float4 k = *(const float4*)(F2 + (size_t)(t0 + r) * DIM + c);
    unsigned short* dk = Ks + r * PITCH + c;
    dk[0] = f2bf(k.x); dk[1] = f2bf(k.y); dk[2] = f2bf(k.z); dk[3] = f2bf(k.w);
  }
  __syncthreads();

  const int wave  = threadIdx.x >> 5;
  const int mBase = wave * 16;
  v8f zero = {0.f, 0.f, 0.f, 0.f, 0.f, 0.f, 0.f, 0.f};
  v8f acc[4] = {zero, zero, zero, zero};

#pragma unroll
  for (int kc = 0; kc < DIM; kc += 32) {
    v16bf a = load_frag_a(Qs, PITCH, mBase, kc);
#pragma unroll
    for (int g = 0; g < 4; ++g) {
      v16bf bb = load_frag_b(Ks, PITCH, g * 16, kc);
      acc[g] = wmma_bf16(a, bb, acc[g]);
    }
  }

  const int lane = threadIdx.x & 31;
  const int n = lane & 15, h = lane >> 4;

  if (mode == 0) {
#pragma unroll
    for (int g = 0; g < 4; ++g) {            // column (t) max over this wave's 16 rows
      float cm = acc[g][0];
#pragma unroll
      for (int i = 1; i < 8; ++i) cm = fmaxf(cm, acc[g][i]);
      cm = fmaxf(cm, __shfl_xor(cm, 16, 32));
      if (h == 0) atomicMax(&colmaxE[bS + t0 + g * 16 + n], fenc(cm));
    }
#pragma unroll
    for (int i = 0; i < 8; ++i) {            // row (s) max over 64 columns
      float rm = acc[0][i];
#pragma unroll
      for (int g = 1; g < 4; ++g) rm = fmaxf(rm, acc[g][i]);
#pragma unroll
      for (int off = 8; off; off >>= 1) rm = fmaxf(rm, __shfl_xor(rm, off, 32));
      if (n == 0) atomicMax(&rowmaxE[bS + s0 + mBase + i + 8 * h], fenc(rm));
    }
  } else {
#pragma unroll
    for (int g = 0; g < 4; ++g) {
      float mx = fdec(colmaxE[bS + t0 + g * 16 + n]);
      float cs = 0.f;
#pragma unroll
      for (int i = 0; i < 8; ++i) cs += __expf(acc[g][i] - mx);
      cs += __shfl_xor(cs, 16, 32);
      if (h == 0) atomicAdd(&colsum[bS + t0 + g * 16 + n], cs);
    }
#pragma unroll
    for (int i = 0; i < 8; ++i) {
      int srow = s0 + mBase + i + 8 * h;
      float mx = fdec(rowmaxE[bS + srow]);
      float rs = 0.f;
#pragma unroll
      for (int g = 0; g < 4; ++g) rs += __expf(acc[g][i] - mx);
#pragma unroll
      for (int off = 8; off; off >>= 1) rs += __shfl_xor(rs, off, 32);
      if (n == 0) atomicAdd(&rowsum[bS + srow], rs);
    }
  }
}

// Block (8 waves) owns 128 output rows (from fOwn), streams 64-wide tiles of fIter:
//   M   = fOwn @ fIter^T              (WMMA, bf16)
//   P   = exp(M - max[t]) / sum[t]    (stats indexed by the iterated axis)
//   O  += P @ fIter                   (WMMA, bf16, 256-wide accumulator in regs)
// Epilogue: out[b,row, colOffset+d] = O[row,d] * fOwn[row,d]
__global__ void __launch_bounds__(256) attn_kernel(
    const float* __restrict__ fOwn, const float* __restrict__ fIter,
    const unsigned int* __restrict__ maxE, const float* __restrict__ sumv,
    float* __restrict__ out, int colOffset)
{
  extern __shared__ unsigned short sm[];
  unsigned short* Qs  = sm;                      // [128][PITCH]  fOwn strip [s][d]
  unsigned short* Ktd = sm + 128 * PITCH;        // [64][PITCH]   fIter tile [t][d]
  unsigned short* Kdt = Ktd + 64 * PITCH;        // [DIM][PPITCH] fIter tile [d][t]
  unsigned short* Ps  = Kdt + DIM * PPITCH;      // [128][PPITCH] P tile [s][t]

  const int b  = blockIdx.y;
  const int s0 = blockIdx.x * 128;
  const size_t bS = (size_t)b * SEQ;
  const float* FO = fOwn  + bS * DIM;
  const float* FI = fIter + bS * DIM;
  const unsigned int* mE = maxE + bS;
  const float* sv = sumv + bS;

  for (int i = threadIdx.x; i < 128 * 64; i += 256) {
    int r = i >> 6, c = (i & 63) * 4;
    float4 q = *(const float4*)(FO + (size_t)(s0 + r) * DIM + c);
    unsigned short* dq = Qs + r * PITCH + c;
    dq[0] = f2bf(q.x); dq[1] = f2bf(q.y); dq[2] = f2bf(q.z); dq[3] = f2bf(q.w);
  }

  const int wave  = threadIdx.x >> 5;
  const int mBase = wave * 16;
  const int lane  = threadIdx.x & 31;
  const int n = lane & 15, h = lane >> 4;

  v8f zero = {0.f, 0.f, 0.f, 0.f, 0.f, 0.f, 0.f, 0.f};
  v8f o[16];
#pragma unroll
  for (int g = 0; g < 16; ++g) o[g] = zero;

  for (int t0 = 0; t0 < SEQ; t0 += 64) {
    const bool hasNext = (t0 + 64) < SEQ;
    const float* nextFI = FI + (size_t)(t0 + 64) * DIM;

    __syncthreads();   // prev PV done reading Ktd/Kdt/Ps (also covers Qs fill)
    for (int i = threadIdx.x; i < 64 * 64; i += 256) {
      int r = i >> 6, c = (i & 63) * 4;
      float4 v = *(const float4*)(FI + (size_t)(t0 + r) * DIM + c);
      if (hasNext)  // pull next tile toward cache while WMMA runs (global_prefetch_b8)
        __builtin_prefetch(nextFI + (size_t)r * DIM + c, 0, 1);
      unsigned short b0 = f2bf(v.x), b1 = f2bf(v.y), b2 = f2bf(v.z), b3 = f2bf(v.w);
      unsigned short* dt = Ktd + r * PITCH + c;
      dt[0] = b0; dt[1] = b1; dt[2] = b2; dt[3] = b3;
      Kdt[(c + 0) * PPITCH + r] = b0;
      Kdt[(c + 1) * PPITCH + r] = b1;
      Kdt[(c + 2) * PPITCH + r] = b2;
      Kdt[(c + 3) * PPITCH + r] = b3;
    }
    __syncthreads();

    v8f macc[4] = {zero, zero, zero, zero};
#pragma unroll
    for (int kc = 0; kc < DIM; kc += 32) {
      v16bf a = load_frag_a(Qs, PITCH, mBase, kc);
#pragma unroll
      for (int g = 0; g < 4; ++g) {
        v16bf bb = load_frag_b(Ktd, PITCH, g * 16, kc);
        macc[g] = wmma_bf16(a, bb, macc[g]);
      }
    }

#pragma unroll
    for (int g = 0; g < 4; ++g) {
      int t = t0 + g * 16 + n;
      float mx  = fdec(mE[t]);
      float inv = 1.0f / sv[t];
#pragma unroll
      for (int i = 0; i < 8; ++i) {
        float p = __expf(macc[g][i] - mx) * inv;
        Ps[(mBase + i + 8 * h) * PPITCH + g * 16 + n] = f2bf(p);
      }
    }
    __syncthreads();

#pragma unroll
    for (int kc = 0; kc < 64; kc += 32) {
      v16bf a = load_frag_a(Ps, PPITCH, mBase, kc);
#pragma unroll
      for (int g = 0; g < 16; ++g) {
        v16bf bb = load_frag_b(Kdt, PPITCH, g * 16, kc);
        o[g] = wmma_bf16(a, bb, o[g]);
      }
    }
  }

#pragma unroll
  for (int g = 0; g < 16; ++g) {
#pragma unroll
    for (int i = 0; i < 8; ++i) {
      int row = s0 + mBase + i + 8 * h;
      int d   = g * 16 + n;
      float val = o[g][i] * FO[(size_t)row * DIM + d];
      out[(bS + row) * (size_t)(2 * DIM) + colOffset + d] = val;
    }
  }
}

extern "C" void kernel_launch(void* const* d_in, const int* in_sizes, int n_in,
                              void* d_out, int out_size, void* d_ws, size_t ws_size,
                              hipStream_t stream) {
  (void)in_sizes; (void)n_in; (void)out_size; (void)ws_size;

  const float* emb[3] = {(const float*)d_in[0], (const float*)d_in[1],
                         (const float*)d_in[2]};
  const int pairs[3][2] = {{0, 1}, {0, 2}, {1, 2}};   // (a,v), (a,l), (v,l)
  float* out = (float*)d_out;
  unsigned int* ws = (unsigned int*)d_ws;
  const size_t BS = (size_t)BATCH * SEQ;

  hipFuncSetAttribute(reinterpret_cast<const void*>(stats_kernel),
                      hipFuncAttributeMaxDynamicSharedMemorySize, STATS_SMEM);
  hipFuncSetAttribute(reinterpret_cast<const void*>(attn_kernel),
                      hipFuncAttributeMaxDynamicSharedMemorySize, ATTN_SMEM);

  {
    size_t nInit = (size_t)3 * 4 * BS;
    init_ws<<<dim3((unsigned)((nInit + 255) / 256)), 256, 0, stream>>>(ws, nInit);
  }

  dim3 gs(SEQ / 128, SEQ / 64, BATCH);
  for (int p = 0; p < 3; ++p) {
    unsigned int* cmE = ws + (size_t)(p * 4 + 0) * BS;
    float*        cs  = (float*)(ws + (size_t)(p * 4 + 1) * BS);
    unsigned int* rmE = ws + (size_t)(p * 4 + 2) * BS;
    float*        rs  = (float*)(ws + (size_t)(p * 4 + 3) * BS);
    stats_kernel<<<gs, 256, STATS_SMEM, stream>>>(emb[pairs[p][0]], emb[pairs[p][1]],
                                                  cmE, rmE, cs, rs, 0);
  }
  for (int p = 0; p < 3; ++p) {
    unsigned int* cmE = ws + (size_t)(p * 4 + 0) * BS;
    float*        cs  = (float*)(ws + (size_t)(p * 4 + 1) * BS);
    unsigned int* rmE = ws + (size_t)(p * 4 + 2) * BS;
    float*        rs  = (float*)(ws + (size_t)(p * 4 + 3) * BS);
    stats_kernel<<<gs, 256, STATS_SMEM, stream>>>(emb[pairs[p][0]], emb[pairs[p][1]],
                                                  cmE, rmE, cs, rs, 1);
  }

  dim3 ga(SEQ / 128, BATCH);
  for (int p = 0; p < 3; ++p) {
    unsigned int* cmE = ws + (size_t)(p * 4 + 0) * BS;
    float*        cs  = (float*)(ws + (size_t)(p * 4 + 1) * BS);
    unsigned int* rmE = ws + (size_t)(p * 4 + 2) * BS;
    float*        rs  = (float*)(ws + (size_t)(p * 4 + 3) * BS);
    float* po = out + (size_t)p * BS * 2 * DIM;
    // O1/A1: rows = feat1 (s), iterate feat2 (t), col-softmax stats, cols [0,D)
    attn_kernel<<<ga, 256, ATTN_SMEM, stream>>>(emb[pairs[p][0]], emb[pairs[p][1]],
                                                cmE, cs, po, 0);
    // O2/A2: rows = feat2 (t), iterate feat1 (s), row-softmax stats, cols [D,2D)
    attn_kernel<<<ga, 256, ATTN_SMEM, stream>>>(emb[pairs[p][1]], emb[pairs[p][0]],
                                                rmE, rs, po, DIM);
  }
}
